// MultiHeadAttention_1614907703493
// MI455X (gfx1250) — compile-verified, matrix operand order checked by reading, not run on hardware
//
#include <hip/hip_runtime.h>

typedef __attribute__((ext_vector_type(16))) _Float16 v16h;
typedef __attribute__((ext_vector_type(8)))  _Float16 v8h;
typedef __attribute__((ext_vector_type(8)))  float    v8f;

#define HIDDEN  2048
#define SEQ     2048
#define BATCH   2
#define NHEADS  32
#define NKV     8
#define HDIM    64
#define MTOT    (BATCH * SEQ)       // 4096 tokens
#define QSTRIDE (NHEADS * HDIM)     // 2048 halves per token in q/ctx
#define KSTRIDE (NKV * HDIM)        // 512 halves per token in k/v

// ---------------------------------------------------------------------------
// Fragment loader: tile stored in LDS as [row][k] (k contiguous), row stride
// `stride` halves (padded, 16B-aligned rows). Per ISA 16-bit A-matrix layout:
// lane L holds row (L&15); k elements are kbase+{0..7} and kbase+16+{0..7}
// with kbase including the half-wave offset ((L>>4)*8). B fragments use the
// same pattern with "row" = N column (tile stored transposed, [n][k]).
// ---------------------------------------------------------------------------
__device__ __forceinline__ v16h load_frag(const _Float16* base, int row_base,
                                          int stride, int k_base, int lane) {
  const _Float16* p = base + (row_base + (lane & 15)) * stride
                           + k_base + ((lane >> 4) << 3);
  v8h lo = *(const v8h*)p;
  v8h hi = *(const v8h*)(p + 16);
  return __builtin_shufflevector(lo, hi, 0, 1, 2, 3, 4, 5, 6, 7,
                                         8, 9, 10, 11, 12, 13, 14, 15);
}

__device__ __forceinline__ v8f wmma16(v16h a, v16h b, v8f c) {
  return __builtin_amdgcn_wmma_f32_16x16x32_f16(false, a, false, b,
                                                (short)0, c, false, false);
}

// ---------------------------------------------------------------------------
// f32 -> f16 convert
// ---------------------------------------------------------------------------
__global__ void cvt_f32_f16(const float* __restrict__ in,
                            _Float16* __restrict__ out, int n) {
  int i = blockIdx.x * blockDim.x + threadIdx.x;
  if (i < n) out[i] = (_Float16)in[i];
}

// ---------------------------------------------------------------------------
// Tiled f16 GEMM: C[M,N] = A[M,K] * B[K,N].  Block tile 128(M) x 128(N),
// BK = 32, 256 threads = 8 waves in a 4(M) x 2(N) grid; each wave owns a
// 32x64 tile (2 A-frags x 4 B-frags -> 8 WMMAs per BK step, 64 VGPRs of
// accumulators).  A staged [m][k], B staged transposed [n][k].
// ---------------------------------------------------------------------------
template <typename OutT>
__global__ __launch_bounds__(256) void gemm_f16(const _Float16* __restrict__ A,
                                                const _Float16* __restrict__ B,
                                                OutT* __restrict__ C,
                                                int M, int N, int K) {
  __shared__ _Float16 ldsA[128 * 40];   // 128 rows, stride 40 (32 + 8 pad)
  __shared__ _Float16 ldsBT[128 * 40];  // 128 cols, stride 40

  const int tid  = threadIdx.x;
  const int lane = tid & 31;
  const int wave = tid >> 5;
  const int bm = blockIdx.y * 128;
  const int bn = blockIdx.x * 128;
  const int wm = (wave >> 1) * 32;   // 0,32,64,96
  const int wn = (wave & 1) * 64;    // 0,64

  v8f acc[2][4] = {};

  for (int k0 = 0; k0 < K; k0 += 32) {
    // A tile: 128x32 = 4096 halves, 256 threads x 2 x v8h
#pragma unroll
    for (int it = 0; it < 2; ++it) {
      int linear = tid + it * 256;       // 0..511
      int row = linear >> 2;             // 128 rows, 4 segs each
      int cb  = (linear & 3) * 8;
      v8h d = *(const v8h*)(A + (size_t)(bm + row) * K + k0 + cb);
      *(v8h*)&ldsA[row * 40 + cb] = d;
    }
    // B tile transposed: 32x128 = 4096 halves
#pragma unroll
    for (int it = 0; it < 2; ++it) {
      int linear = tid + it * 256;       // 0..511
      int kk = linear & 31;
      int nb = (linear >> 5) * 8;        // 0..120
      v8h d = *(const v8h*)(B + (size_t)(k0 + kk) * N + bn + nb);
#pragma unroll
      for (int j = 0; j < 8; ++j) ldsBT[(nb + j) * 40 + kk] = d[j];
    }
    // Prefetch next K-step tiles (global_prefetch_b8): one line per thread.
    if (k0 + 32 < K) {
      __builtin_prefetch(A + (size_t)(bm + (tid >> 1)) * K + k0 + 32 + (tid & 1) * 16, 0, 1);
      __builtin_prefetch(B + (size_t)(k0 + 32 + (tid & 31)) * N + bn + (tid >> 5) * 16, 0, 1);
    }
    __syncthreads();

    v16h a0 = load_frag(ldsA, wm, 40, 0, lane);
    v16h a1 = load_frag(ldsA, wm + 16, 40, 0, lane);
    v16h bf[4];
#pragma unroll
    for (int j = 0; j < 4; ++j) bf[j] = load_frag(ldsBT, wn + j * 16, 40, 0, lane);
#pragma unroll
    for (int j = 0; j < 4; ++j) {
      acc[0][j] = wmma16(a0, bf[j], acc[0][j]);
      acc[1][j] = wmma16(a1, bf[j], acc[1][j]);
    }
    __syncthreads();
  }

  // Epilogue: C layout — VGPR r holds row r + (lane>>4)*8, col = lane&15
  const int hb = (lane >> 4) << 3;
  const int ln = lane & 15;
#pragma unroll
  for (int i = 0; i < 2; ++i)
#pragma unroll
    for (int j = 0; j < 4; ++j)
#pragma unroll
      for (int r = 0; r < 8; ++r) {
        int row = bm + wm + i * 16 + r + hb;
        int col = bn + wn + j * 16 + ln;
        C[(size_t)row * N + col] = (OutT)acc[i][j][r];
      }
}

// ---------------------------------------------------------------------------
// RoPE in place on q (32 heads) and k (8 heads), f16 storage, f32 math.
// One thread per (token, head, i<32) rotation pair.
// ---------------------------------------------------------------------------
__global__ void rope_kernel(_Float16* __restrict__ q, _Float16* __restrict__ k,
                            const float* __restrict__ cosb,
                            const float* __restrict__ sinb) {
  int gid = blockIdx.x * blockDim.x + threadIdx.x;  // MTOT*40*32 threads
  int i  = gid & 31;
  int hh = (gid >> 5) % 40;
  int t  = gid / (32 * 40);
  int s  = t & (SEQ - 1);
  float c  = cosb[s * 32 + i];
  float sn = sinb[s * 32 + i];
  _Float16* base = (hh < 32) ? (q + (size_t)t * QSTRIDE + hh * HDIM)
                             : (k + (size_t)t * KSTRIDE + (hh - 32) * HDIM);
  float x1 = (float)base[i];
  float x2 = (float)base[i + 32];
  base[i]      = (_Float16)(x1 * c - x2 * sn);
  base[i + 32] = (_Float16)(x2 * c + x1 * sn);
}

// ---------------------------------------------------------------------------
// Flash attention. Grid: B * NHEADS * (SEQ/64) blocks; 128 threads (4 waves).
// Each wave owns 16 query rows. Keys processed in 64-row blocks with online
// softmax; S = Q*K^T and O += P*V both via v_wmma_f32_16x16x32_f16.
// ---------------------------------------------------------------------------
__global__ __launch_bounds__(128) void attn_kernel(const _Float16* __restrict__ q,
                                                   const _Float16* __restrict__ k,
                                                   const _Float16* __restrict__ v,
                                                   _Float16* __restrict__ ctx) {
  __shared__ _Float16 ldsQ[64 * 72];   // [qrow][d]      (Q pre-scaled)
  __shared__ _Float16 ldsK[64 * 72];   // [key][d]
  __shared__ _Float16 ldsVT[64 * 72];  // [d][key]
  __shared__ _Float16 ldsP[4][16 * 72];// per-wave P tile [qrow][key]

  const int tid  = threadIdx.x;
  const int lane = tid & 31;
  const int wave = tid >> 5;
  const int bid = blockIdx.x;
  const int qb  = bid & 31;           // query block (S/64 = 32)
  const int h   = (bid >> 5) & 31;    // head
  const int b   = bid >> 10;          // batch
  const int kvh = h >> 2;             // GQA: 4 heads share a kv head

  const _Float16* qbase = q + ((size_t)(b * SEQ + qb * 64)) * QSTRIDE + h * HDIM;

  // Load + scale Q (64x64), 128 threads x 4 iters x 8 halves
#pragma unroll
  for (int it = 0; it < 4; ++it) {
    int linear = tid + it * 128;            // 0..511
    int row = linear >> 3;
    int cb  = (linear & 7) * 8;
    v8h d = *(const v8h*)(qbase + (size_t)row * QSTRIDE + cb);
#pragma unroll
    for (int j = 0; j < 8; ++j)
      ldsQ[row * 72 + cb + j] = (_Float16)((float)d[j] * 0.125f);  // d^-0.5
  }

  const int hb = (lane >> 4) << 3;
  const int ln = lane & 15;

  float m_run[8], l_run[8];
  v8f o[4] = {};
#pragma unroll
  for (int r = 0; r < 8; ++r) { m_run[r] = -1e30f; l_run[r] = 0.0f; }

  for (int kb = 0; kb <= qb; ++kb) {
    __syncthreads();
    const _Float16* kptr = k + ((size_t)(b * SEQ + kb * 64)) * KSTRIDE + kvh * HDIM;
    const _Float16* vptr = v + ((size_t)(b * SEQ + kb * 64)) * KSTRIDE + kvh * HDIM;
#pragma unroll
    for (int it = 0; it < 4; ++it) {
      int linear = tid + it * 128;
      int row = linear >> 3;
      int cb  = (linear & 7) * 8;
      v8h dk = *(const v8h*)(kptr + (size_t)row * KSTRIDE + cb);
      *(v8h*)&ldsK[row * 72 + cb] = dk;
      v8h dv = *(const v8h*)(vptr + (size_t)row * KSTRIDE + cb);
#pragma unroll
      for (int j = 0; j < 8; ++j) ldsVT[(cb + j) * 72 + row] = dv[j];
    }
    // Prefetch next key block while this one is consumed.
    if (kb < qb) {
      const _Float16* nkp = kptr + (size_t)64 * KSTRIDE;
      const _Float16* nvp = vptr + (size_t)64 * KSTRIDE;
      int prow = tid & 63;
      if (tid < 64) __builtin_prefetch(nkp + (size_t)prow * KSTRIDE, 0, 1);
      else          __builtin_prefetch(nvp + (size_t)prow * KSTRIDE, 0, 1);
    }
    __syncthreads();

    // S = Q * K^T  (16 q-rows x 64 keys per wave)
    v8f sfrag[4];
#pragma unroll
    for (int nt = 0; nt < 4; ++nt) {
      v8f s = {};
#pragma unroll
      for (int dk = 0; dk < 64; dk += 32) {
        v16h af = load_frag(ldsQ, wave * 16, 72, dk, lane);
        v16h bf = load_frag(ldsK, nt * 16, 72, dk, lane);
        s = wmma16(af, bf, s);
      }
      sfrag[nt] = s;
    }

    // causal mask on the diagonal block
    if (kb == qb) {
#pragma unroll
      for (int nt = 0; nt < 4; ++nt)
#pragma unroll
        for (int r = 0; r < 8; ++r) {
          int key  = nt * 16 + ln;
          int qrow = wave * 16 + r + hb;
          if (key > qrow) sfrag[nt][r] = -1e30f;
        }
    }

    // online softmax (row reductions across 16-lane halves)
    float corr[8];
#pragma unroll
    for (int r = 0; r < 8; ++r) {
      float mx = sfrag[0][r];
#pragma unroll
      for (int nt = 1; nt < 4; ++nt) mx = fmaxf(mx, sfrag[nt][r]);
#pragma unroll
      for (int off = 8; off >= 1; off >>= 1)
        mx = fmaxf(mx, __shfl_xor(mx, off, 16));
      float mnew = fmaxf(m_run[r], mx);
      corr[r] = __expf(m_run[r] - mnew);
      float lsum = 0.0f;
#pragma unroll
      for (int nt = 0; nt < 4; ++nt) {
        float p = __expf(sfrag[nt][r] - mnew);
        sfrag[nt][r] = p;
        lsum += p;
      }
#pragma unroll
      for (int off = 8; off >= 1; off >>= 1)
        lsum += __shfl_xor(lsum, off, 16);
      l_run[r] = l_run[r] * corr[r] + lsum;
      m_run[r] = mnew;
    }
#pragma unroll
    for (int dt = 0; dt < 4; ++dt)
#pragma unroll
      for (int r = 0; r < 8; ++r) o[dt][r] *= corr[r];

    // stage P (f16) per wave: C layout -> [qrow][key]
#pragma unroll
    for (int nt = 0; nt < 4; ++nt)
#pragma unroll
      for (int r = 0; r < 8; ++r)
        ldsP[wave][(r + hb) * 72 + nt * 16 + ln] = (_Float16)sfrag[nt][r];
    __syncthreads();

    // O += P * V
#pragma unroll
    for (int dt = 0; dt < 4; ++dt) {
      v8f acc = o[dt];
#pragma unroll
      for (int ks = 0; ks < 64; ks += 32) {
        v16h af = load_frag(&ldsP[wave][0], 0, 72, ks, lane);
        v16h bf = load_frag(ldsVT, dt * 16, 72, ks, lane);
        acc = wmma16(af, bf, acc);
      }
      o[dt] = acc;
    }
  }

  // normalize and store context (token, head, d) f16
  _Float16* cbase = ctx + ((size_t)(b * SEQ + qb * 64 + wave * 16)) * QSTRIDE + h * HDIM;
#pragma unroll
  for (int dt = 0; dt < 4; ++dt)
#pragma unroll
    for (int r = 0; r < 8; ++r) {
      int row = r + hb;
      int col = dt * 16 + ln;
      cbase[(size_t)row * QSTRIDE + col] = (_Float16)(o[dt][r] / l_run[r]);
    }
}

// ---------------------------------------------------------------------------
// Host launcher
// ---------------------------------------------------------------------------
extern "C" void kernel_launch(void* const* d_in, const int* in_sizes, int n_in,
                              void* d_out, int out_size, void* d_ws, size_t ws_size,
                              hipStream_t stream) {
  (void)in_sizes; (void)n_in; (void)out_size; (void)ws_size;

  const float* x_f   = (const float*)d_in[0];
  const float* cos_f = (const float*)d_in[1];
  const float* sin_f = (const float*)d_in[2];
  const float* Wq_f  = (const float*)d_in[3];
  const float* Wk_f  = (const float*)d_in[4];
  const float* Wv_f  = (const float*)d_in[5];
  const float* Wo_f  = (const float*)d_in[6];
  float* out = (float*)d_out;

  char* ws = (char*)d_ws;
  size_t off = 0;
  auto alloc = [&](size_t bytes) -> _Float16* {
    _Float16* p = (_Float16*)(ws + off);
    off += (bytes + 255) & ~(size_t)255;
    return p;
  };
  _Float16* xh  = alloc((size_t)MTOT * HIDDEN * 2);
  _Float16* Wqh = alloc((size_t)HIDDEN * QSTRIDE * 2);
  _Float16* Wkh = alloc((size_t)HIDDEN * KSTRIDE * 2);
  _Float16* Wvh = alloc((size_t)HIDDEN * KSTRIDE * 2);
  _Float16* Woh = alloc((size_t)QSTRIDE * HIDDEN * 2);
  _Float16* qb  = alloc((size_t)MTOT * QSTRIDE * 2);
  _Float16* kb  = alloc((size_t)MTOT * KSTRIDE * 2);
  _Float16* vb  = alloc((size_t)MTOT * KSTRIDE * 2);
  _Float16* ctx = alloc((size_t)MTOT * QSTRIDE * 2);

  auto cvt = [&](const float* src, _Float16* dst, int n) {
    cvt_f32_f16<<<(n + 255) / 256, 256, 0, stream>>>(src, dst, n);
  };
  cvt(x_f,  xh,  MTOT * HIDDEN);
  cvt(Wq_f, Wqh, HIDDEN * QSTRIDE);
  cvt(Wk_f, Wkh, HIDDEN * KSTRIDE);
  cvt(Wv_f, Wvh, HIDDEN * KSTRIDE);
  cvt(Wo_f, Woh, QSTRIDE * HIDDEN);

  // QKV projections
  {
    dim3 gq(QSTRIDE / 128, MTOT / 128);
    gemm_f16<_Float16><<<gq, 256, 0, stream>>>(xh, Wqh, qb, MTOT, QSTRIDE, HIDDEN);
    dim3 gkv(KSTRIDE / 128, MTOT / 128);
    gemm_f16<_Float16><<<gkv, 256, 0, stream>>>(xh, Wkh, kb, MTOT, KSTRIDE, HIDDEN);
    gemm_f16<_Float16><<<gkv, 256, 0, stream>>>(xh, Wvh, vb, MTOT, KSTRIDE, HIDDEN);
  }

  // RoPE on q and k
  {
    int total = MTOT * 40 * 32;
    rope_kernel<<<total / 256, 256, 0, stream>>>(qb, kb, cos_f, sin_f);
  }

  // Flash attention
  attn_kernel<<<BATCH * NHEADS * (SEQ / 64), 128, 0, stream>>>(qb, kb, vb, ctx);

  // Output projection -> f32
  {
    dim3 go(HIDDEN / 128, MTOT / 128);
    gemm_f16<float><<<go, 256, 0, stream>>>(ctx, Woh, out, MTOT, HIDDEN, QSTRIDE);
  }
}
